// Attention_43954695307588
// MI455X (gfx1250) — matrix operand order repaired and checked by reading, not compiled, not run on hardware
//
#include <hip/hip_runtime.h>
#include <stdint.h>

// ---------------------------------------------------------------------------
// GQA decode attention for MI455X (gfx1250, wave32).
// Memory-bound (~500MB streamed / ~3.7 GFLOP): kernels stream weights and the
// int8 KV cache exactly once; WMMA does the (cheap) math; async-to-LDS copies
// and global prefetch hide latency on the streams.
// ---------------------------------------------------------------------------

typedef __attribute__((ext_vector_type(2)))  float    v2f;
typedef __attribute__((ext_vector_type(8)))  float    v8f;
typedef __attribute__((ext_vector_type(16))) _Float16 v16h;

#define B_    32
#define H_    4096
#define NH_   32
#define NKV_  8
#define HD_   128
#define PAST_ 4096
#define QKVN_ ((NH_ + 2 * NKV_) * HD_)   // 6144
#define TS_   4112                        // padded score row stride (>4097)

union H16 { v16h v; _Float16 e[16]; };

__device__ __forceinline__ float b2f(unsigned w, int i) {
  // sign-extend byte i of w to int, convert to float
  return (float)((int)(w << (24 - 8 * i)) >> 24);
}

// Dequantize 16 contiguous int8 (two G=8 quant groups) into 16 f16 lanes.
__device__ __forceinline__ void dq16(const int8_t* p, float s0, float s1,
                                     _Float16* e) {
  const uint4 raw = *(const uint4*)p;
#pragma unroll
  for (int i = 0; i < 4; ++i) {
    e[i]      = (_Float16)(b2f(raw.x, i) * s0);
    e[4 + i]  = (_Float16)(b2f(raw.y, i) * s0);
    e[8 + i]  = (_Float16)(b2f(raw.z, i) * s1);
    e[12 + i] = (_Float16)(b2f(raw.w, i) * s1);
  }
}

// Async copy of 16 bytes global -> LDS (per lane), tracked by ASYNCcnt.
__device__ __forceinline__ void async_g2l_b128(uint32_t lds_byte_off,
                                               const void* gaddr) {
  asm volatile("global_load_async_to_lds_b128 %0, %1, off"
               :: "v"(lds_byte_off), "v"(gaddr)
               : "memory");
}
__device__ __forceinline__ void wait_asynccnt0() {
  asm volatile("s_wait_asynccnt 0x0" ::: "memory");
}

// ---------------------------------------------------------------------------
// Kernel 1/4: C[32,N] = A[32,K] @ B[K,N], fp32 WMMA 16x16x4.
// grid = (N/64, KSPLIT), block = 256 (8 waves: 2 M-tiles x 4 N-tiles).
// A chunk staged in LDS via async-to-LDS b128; B streamed with SADDR-form
// 32-bit offsets + next-chunk prefetch. Deterministic split-K partials.
// ---------------------------------------------------------------------------
__global__ __launch_bounds__(256)
void gemm32_f32wmma(const float* __restrict__ A, const float* __restrict__ Bm,
                    float* __restrict__ Cpart, int K, int N) {
  constexpr int KC = 256;
  __shared__ float sA[32 * KC];          // 32KB: A chunk, reused by all waves
  const int tid = threadIdx.x;
  const int w = tid >> 5, lane = tid & 31, lm = lane & 15, hi = lane >> 4;
  const int m0 = (w >> 2) * 16;
  const int n0 = blockIdx.x * 64 + (w & 3) * 16;
  const int kchunk = K / gridDim.y;
  const int kbeg = blockIdx.y * kchunk;
  const uint32_t sAoff = (uint32_t)(uintptr_t)&sA[0];  // LDS byte offset

  v8f c = {};
  for (int k0 = kbeg; k0 < kbeg + kchunk; k0 += KC) {
    // async-stage A[0:32, k0:k0+KC) into LDS: 8 x (256 lanes x 16B)
#pragma unroll
    for (int it = 0; it < (32 * KC) / (256 * 4); ++it) {
      const int idx = it * 1024 + tid * 4;           // float index into sA
      const int r = idx >> 8, cc = idx & (KC - 1);
      async_g2l_b128(sAoff + (uint32_t)idx * 4u,
                     A + (size_t)((unsigned)(r * K + k0 + cc)));
    }
    wait_asynccnt0();
    __syncthreads();

#pragma unroll 4
    for (int kk = 0; kk < KC; kk += 4) {
      const int krel = kk + 2 * hi;      // A/B 32-bit frag: K = 2*hi + r
      v2f a, b;
      a.x = sA[(m0 + lm) * KC + krel];
      a.y = sA[(m0 + lm) * KC + krel + 1];
      const unsigned boff = (unsigned)((k0 + krel) * N + n0 + lm);
      b.x = Bm[boff];
      b.y = Bm[boff + (unsigned)N];
      if (k0 + 2 * KC <= K)              // prefetch next chunk's row
        __builtin_prefetch(Bm + boff + (size_t)KC * N, 0, 0);
      c = __builtin_amdgcn_wmma_f32_16x16x4_f32(false, a, false, b, (short)0,
                                                c, false, false);
    }
    __syncthreads();
  }
  // C frag: VGPR r holds row m0 + r + 8*hi, col n0 + lm
  float* Cp = Cpart + (size_t)blockIdx.y * 32 * N;
#pragma unroll
  for (int r = 0; r < 8; ++r)
    Cp[(size_t)(m0 + r + 8 * hi) * N + n0 + lm] = c[r];
}

__global__ __launch_bounds__(256)
void reduce_splitk(const float* __restrict__ part, float* __restrict__ C,
                   int total, int nsplit) {
  int i = blockIdx.x * 256 + threadIdx.x;
  if (i >= total) return;
  float s = 0.f;
  for (int y = 0; y < nsplit; ++y) s += part[(size_t)y * total + i];
  C[i] = s;
}

// ---------------------------------------------------------------------------
// Kernel 2: RoPE (half-split) on q heads (0..31) and k heads (32..39),
// in place on the qkv workspace [B, 48, 128]. S == 1 -> single position.
// ---------------------------------------------------------------------------
__global__ __launch_bounds__(64)
void rope_kernel(float* __restrict__ qkv, const int* __restrict__ start_pos) {
  const int b = blockIdx.x / 40;
  const int h = blockIdx.x % 40;
  const int j = threadIdx.x;             // 0..63 = half of HD
  const float pos = (float)(*start_pos);
  const float inv = __expf(-(float)j * (logf(10000.0f) / 64.0f));
  const float ang = pos * inv;
  float s, c;
  __sincosf(ang, &s, &c);
  float* base = qkv + ((size_t)b * 48 + h) * HD_;
  const float t1 = base[j], t2 = base[j + 64];
  base[j]      = t1 * c - t2 * s;
  base[j + 64] = t2 * c + t1 * s;
}

// ---------------------------------------------------------------------------
// Kernel 3: flash decode. One block per (b, kv-head g). 8 waves.
// Pass 1: scores[4, 4096] via f16 WMMA over dequantized K tiles -> LDS.
//   A (q, loop-invariant): 16-bit A frag, same K = hi*16+j order as B.
//   B (K-cache tile):      16-bit B frag, this code's K = hi*16+j  -> one 16B load.
// Tail score for the freshly-computed k (t = 4096) done in fp32.
// Softmax per head by one wave (shfl reductions), p overwrites scores in LDS.
// Pass 2: out[4,128] = p @ V via f16 WMMA; wave w owns dims [16w,16w+16).
// ---------------------------------------------------------------------------
__global__ __launch_bounds__(256)
void attn_decode(const float* __restrict__ qkvb,      // [B,48,128] (post-RoPE)
                 const int8_t* __restrict__ kvc,      // [2,B,8,4096,128]
                 const float* __restrict__ kvs,       // [2,B,8,4096,16]
                 float* __restrict__ attnOut)         // [B, 32*128]
{
  extern __shared__ char smem[];
  float* sQ   = (float*)smem;            // 4*128
  float* sXV  = sQ + 4 * HD_;            // 128
  float* sRed = sXV + HD_;               // 16 (inv_l per head)
  float* sS   = sRed + 16;               // 4 * TS_ (scores -> probs)

  const int b = blockIdx.x >> 3;
  const int g = blockIdx.x & 7;
  const int tid = threadIdx.x;
  const int w = tid >> 5, lane = tid & 31, lm = lane & 15, hi = lane >> 4;

  const float* qkvB = qkvb + (size_t)b * 48 * HD_;
  for (int i = tid; i < 4 * HD_; i += 256) {
    int m = i >> 7, d = i & 127;
    sQ[i] = qkvB[(g * 4 + m) * HD_ + d];
  }
  if (tid < HD_) sXV[tid] = qkvB[(40 + g) * HD_ + tid];
  __syncthreads();

  const int8_t* Kc = kvc + (size_t)(((size_t)b) * NKV_ + g) * PAST_ * HD_;
  const int8_t* Vc = Kc + (size_t)B_ * NKV_ * PAST_ * HD_;
  const float*  Ks = kvs + (size_t)(((size_t)b) * NKV_ + g) * PAST_ * 16;
  const float*  Vs = Ks + (size_t)B_ * NKV_ * PAST_ * 16;

  // Loop-invariant q fragments: 4 chunks of 32 head-dims (rows 4..15 zero).
  H16 aq[4];
#pragma unroll
  for (int cidx = 0; cidx < 4; ++cidx) {
    const int kc = cidx * 32;
#pragma unroll
    for (int j = 0; j < 16; ++j) {
      const int kk = kc + hi * 16 + j;
      const float v = (lm < 4) ? sQ[lm * HD_ + kk] : 0.f;
      aq[cidx].e[j] = (_Float16)v;
    }
  }

  const float rscale = 0.08838834764831845f;   // 1/sqrt(128)

  // ---- Pass 1: scores over the int8 K cache --------------------------------
  for (int t0 = w * 16; t0 < PAST_; t0 += 8 * 16) {
    v8f c = {};
    const int t = t0 + lm;                      // token for this lane's column
    const int8_t* krow = Kc + (size_t)t * HD_;
    const float*  srow = Ks + (size_t)t * 16;
    if (t0 + 160 <= PAST_) {                    // next tile for this wave
      __builtin_prefetch(krow + 128 * HD_, 0, 0);
      __builtin_prefetch(srow + 128 * 16, 0, 0);
    }
#pragma unroll
    for (int cidx = 0; cidx < 4; ++cidx) {
      const int d0 = cidx * 32 + hi * 16;       // B frag: K = hi*16 + j
      H16 bk;
      dq16(krow + d0, srow[d0 >> 3], srow[(d0 >> 3) + 1], bk.e);
      c = __builtin_amdgcn_wmma_f32_16x16x32_f16(false, aq[cidx].v, false,
                                                 bk.v, (short)0, c, false,
                                                 false);
    }
    if (hi == 0) {                              // rows M=0..3 are real heads
#pragma unroll
      for (int r = 0; r < 4; ++r) sS[r * TS_ + t] = c[r] * rscale;
    }
  }
  // current-token score (t = PAST_) with the fp32 roped k
  if (w == 0) {
    const float* knew = qkvB + (32 + g) * HD_;
#pragma unroll
    for (int h = 0; h < 4; ++h) {
      float p = 0.f;
      for (int d = lane; d < HD_; d += 32) p += sQ[h * HD_ + d] * knew[d];
      for (int off = 16; off; off >>= 1) p += __shfl_xor(p, off, 32);
      if (lane == 0) sS[h * TS_ + PAST_] = p * rscale;
    }
  }
  __syncthreads();

  // ---- Softmax: one wave per head -----------------------------------------
  if (w < 4) {
    const int h = w;
    float m = -1e30f;
    for (int t = lane; t < PAST_ + 1; t += 32) m = fmaxf(m, sS[h * TS_ + t]);
    for (int off = 16; off; off >>= 1) m = fmaxf(m, __shfl_xor(m, off, 32));
    float l = 0.f;
    for (int t = lane; t < PAST_ + 1; t += 32) {
      const float p = __expf(sS[h * TS_ + t] - m);
      sS[h * TS_ + t] = p;
      l += p;
    }
    for (int off = 16; off; off >>= 1) l += __shfl_xor(l, off, 32);
    if (lane == 0) sRed[h] = 1.f / l;
  }
  __syncthreads();

  // ---- Pass 2: out = p @ V. Wave w owns head-dims [16w, 16w+16). ----------
  const int n0 = w * 16;
  const int dcol = n0 + lm;
  v8f co = {};
  for (int t0 = 0; t0 < PAST_; t0 += 32) {
    if (t0 + 64 <= PAST_) {                     // next tile's V rows + scales
      __builtin_prefetch(Vc + (size_t)(t0 + 32 + hi * 16 + lm) * HD_, 0, 0);
      __builtin_prefetch(Vs + (size_t)(t0 + 32 + lane) * 16, 0, 0);
    }
    H16 ap, bv;
#pragma unroll
    for (int j = 0; j < 16; ++j) {              // A frag: p rows = heads
      const int t = t0 + hi * 16 + j;
      ap.e[j] = (_Float16)((lm < 4) ? sS[lm * TS_ + t] : 0.f);
    }
    const int grp = dcol >> 3;
#pragma unroll
    for (int j = 0; j < 16; ++j) {              // B frag: K tokens = hi*16+j
      const int t = t0 + hi * 16 + j;
      const float sc = Vs[(size_t)t * 16 + grp];
      bv.e[j] = (_Float16)((float)Vc[(size_t)t * HD_ + dcol] * sc);
    }
    co = __builtin_amdgcn_wmma_f32_16x16x32_f16(false, ap.v, false, bv.v,
                                                (short)0, co, false, false);
  }
  // tail token (fp32 v) + normalization + store
  if (hi == 0) {
#pragma unroll
    for (int r = 0; r < 4; ++r) {
      const float val =
          (co[r] + sS[r * TS_ + PAST_] * sXV[dcol]) * sRed[r];
      attnOut[(size_t)b * (NH_ * HD_) + (g * 4 + r) * HD_ + dcol] = val;
    }
  }
}

// ---------------------------------------------------------------------------
extern "C" void kernel_launch(void* const* d_in, const int* in_sizes, int n_in,
                              void* d_out, int out_size, void* d_ws,
                              size_t ws_size, hipStream_t stream) {
  const float*  x         = (const float*)d_in[0];
  const float*  wqkv      = (const float*)d_in[1];
  const float*  wo        = (const float*)d_in[2];
  const int8_t* kvc       = (const int8_t*)d_in[3];
  const float*  kvs       = (const float*)d_in[4];
  const int*    start_pos = (const int*)d_in[5];
  float* out = (float*)d_out;

  const int KSPLIT = 8;
  const size_t qkvN  = (size_t)B_ * QKVN_;        // 196608
  const size_t attnN = (size_t)B_ * NH_ * HD_;    // 131072
  float* qkv   = (float*)d_ws;
  float* attn  = qkv + qkvN;
  float* part1 = attn + attnN;                    // KSPLIT * qkvN
  float* part2 = part1 + (size_t)KSPLIT * qkvN;   // KSPLIT * attnN

  // 1) fused QKV projection (fp32 WMMA, deterministic split-K)
  gemm32_f32wmma<<<dim3(QKVN_ / 64, KSPLIT), 256, 0, stream>>>(
      x, wqkv, part1, H_, QKVN_);
  reduce_splitk<<<((int)qkvN + 255) / 256, 256, 0, stream>>>(
      part1, qkv, (int)qkvN, KSPLIT);

  // 2) RoPE on q + k heads (v untouched)
  rope_kernel<<<B_ * (NH_ + NKV_), 64, 0, stream>>>(qkv, start_pos);

  // 3) flash decode over int8 KV cache (f16 WMMA)
  const size_t smem = (size_t)(4 * HD_ + HD_ + 16 + 4 * TS_) * sizeof(float);
  attn_decode<<<B_ * NKV_, 256, smem, stream>>>(qkv, kvc, kvs, attn);

  // 4) output projection
  gemm32_f32wmma<<<dim3(H_ / 64, KSPLIT), 256, 0, stream>>>(
      attn, wo, part2, NH_ * HD_, H_);
  reduce_splitk<<<((int)attnN + 255) / 256, 256, 0, stream>>>(
      part2, out, (int)attnN, KSPLIT);
}
